// GraphMETNetwork_55482387529804
// MI455X (gfx1250) — matrix operand order repaired and checked by reading, not compiled, not run on hardware
//
#include <hip/hip_runtime.h>

// ---------------------------------------------------------------------------
// GraphMETNetwork on gfx1250 (MI455X).
// Edge message MLP (concat[xi, xj-xi] @ W_msg, 128x64) via
// v_wmma_f32_16x16x32_f16: 16 edges per wave-tile => 4 Ktiles x 4 Ntiles = 16
// WMMAs, bias folded into the WMMA accumulator init. Segment-max via native
// global_atomic_max_num_f32 (no-return) into an L2-resident agg buffer.
// Persistent edge kernel amortizes LDS weight staging; atomic destination
// pointers hoisted into registers so the asm "memory" clobber cannot force
// per-atomic LDS reloads.
// ---------------------------------------------------------------------------

typedef __attribute__((ext_vector_type(16))) _Float16 v16h;
typedef __attribute__((ext_vector_type(8)))  _Float16 v8h;
typedef __attribute__((ext_vector_type(4)))  _Float16 h4;
typedef __attribute__((ext_vector_type(2)))  _Float16 h2;
typedef __attribute__((ext_vector_type(8)))  float    v8f;

#define HDIM 64
#define SAP 136   // padded A-tile row stride in halfs (272B, 16B aligned, bank-spread)
#define SBP 24    // padded B-fragment lane stride in halfs (48B, 16B aligned)
#define EPSV 1e-3f

__device__ __forceinline__ float eluf(float x) {
    return x > 0.0f ? x : (__expf(x) - 1.0f);
}

// Native IEEE maximumNumber float atomic, no return (STOREcnt path), with
// immediate byte offset. agg is initialized to -inf; messages are finite,
// so this is exact segment_max. Deterministic (max is order independent).
__device__ __forceinline__ void atomicMaxFOff(float* addr, float val, int byteOff) {
    asm volatile("global_atomic_max_num_f32 %0, %1, off offset:%2 scope:SCOPE_DEV"
                 :: "v"(addr), "v"(val), "i"(byteOff) : "memory");
}

// ---------------------------------------------------------------------------
// Kernel 1: node encoder.  emb = bn0(elu(concat[emb_cat, emb_cont] @ W_enc))
// One thread per node; weights broadcast from LDS.
// ---------------------------------------------------------------------------
__global__ __launch_bounds__(256) void k_encode(
    const float* __restrict__ x_cont, const int* __restrict__ x_cat,
    const float* __restrict__ datanorm, const float* __restrict__ emb_charge,
    const float* __restrict__ emb_pdg,
    const float* __restrict__ W_cont, const float* __restrict__ b_cont,
    const float* __restrict__ W_cat,  const float* __restrict__ b_cat,
    const float* __restrict__ W_enc,  const float* __restrict__ b_enc,
    const float* __restrict__ bn0_g,  const float* __restrict__ bn0_b,
    const float* __restrict__ bn0_m,  const float* __restrict__ bn0_v,
    float* __restrict__ emb, _Float16* __restrict__ emb_h, int N)
{
    __shared__ float sWc[8 * 32], sbc[32];
    __shared__ float sWk[32 * 32], sbk[32];
    __shared__ float sWe[64 * 64], sbe[64];
    __shared__ float sdn[8], sec[3 * 16], sep[7 * 16];
    __shared__ float sg[64], sb[64], sm[64], sv[64];

    const int tid = threadIdx.x;
    for (int i = tid; i < 8 * 32;  i += 256) sWc[i] = W_cont[i];
    for (int i = tid; i < 32 * 32; i += 256) sWk[i] = W_cat[i];
    for (int i = tid; i < 64 * 64; i += 256) sWe[i] = W_enc[i];
    if (tid < 32) { sbc[tid] = b_cont[tid]; sbk[tid] = b_cat[tid]; }
    if (tid < 64) {
        sbe[tid] = b_enc[tid];
        sg[tid] = bn0_g[tid]; sb[tid] = bn0_b[tid];
        sm[tid] = bn0_m[tid]; sv[tid] = bn0_v[tid];
    }
    if (tid < 8)  sdn[tid] = datanorm[tid];
    if (tid < 48) sec[tid] = emb_charge[tid];
    if (tid < 112) sep[tid] = emb_pdg[tid];
    __syncthreads();

    const int n = blockIdx.x * 256 + tid;
    if (n >= N) return;

    // --- categorical branch -> xin[0:32) ---
    int pdg = x_cat[n * 2 + 0]; if (pdg < 0) pdg = -pdg;
    const int P[7] = {1, 2, 11, 13, 22, 130, 211};
    int vidx = pdg;
    #pragma unroll
    for (int i = 0; i < 7; i++) vidx = (vidx == P[i]) ? i : vidx;
    const int chg = x_cat[n * 2 + 1] + 1;

    float cin[32];
    #pragma unroll
    for (int i = 0; i < 16; i++) {
        cin[i]      = sec[chg * 16 + i];
        cin[16 + i] = sep[vidx * 16 + i];
    }
    float xin[64];
    #pragma unroll
    for (int j = 0; j < 32; j++) {
        float s = sbk[j];
        #pragma unroll
        for (int k = 0; k < 32; k++) s += cin[k] * sWk[k * 32 + j];
        xin[j] = eluf(s);
    }
    // --- continuous branch -> xin[32:64) ---
    float xc[8];
    #pragma unroll
    for (int k = 0; k < 8; k++) xc[k] = x_cont[n * 8 + k] * sdn[k];
    #pragma unroll
    for (int j = 0; j < 32; j++) {
        float s = sbc[j];
        #pragma unroll
        for (int k = 0; k < 8; k++) s += xc[k] * sWc[k * 32 + j];
        xin[32 + j] = eluf(s);
    }
    // --- encoder 64x64 + ELU + BN0 ---
    #pragma unroll
    for (int j = 0; j < 64; j++) {
        float s = sbe[j];
        #pragma unroll
        for (int k = 0; k < 64; k++) s += xin[k] * sWe[k * 64 + j];
        s = eluf(s);
        s = (s - sm[j]) * rsqrtf(sv[j] + EPSV) * sg[j] + sb[j];
        emb[(size_t)n * HDIM + j]   = s;
        emb_h[(size_t)n * HDIM + j] = (_Float16)s;
    }
}

// ---------------------------------------------------------------------------
// Kernel 2: fill agg with -inf bits (128-bit stores).
// ---------------------------------------------------------------------------
__global__ __launch_bounds__(256) void k_fill_neginf(float4* __restrict__ p, size_t n4)
{
    const size_t i = (size_t)blockIdx.x * 256 + threadIdx.x;
    if (i >= n4) return;
    const float ninf = __uint_as_float(0xFF800000u);
    float4 v; v.x = ninf; v.y = ninf; v.z = ninf; v.w = ninf;
    p[i] = v;
}

// ---------------------------------------------------------------------------
// Kernel 3: edge message MLP + segment-max (WMMA), persistent grid.
// Each wave owns tiles of 16 edges: A = [xi | xj-xi] (16x128 f16),
// B = W_msg (128x64 f16, staged once per block), C = 16x64 f32 ->
// global_atomic_max_num_f32 into agg[dst].
// Fragment layouts per CDNA5 ISA 7.12.2 (wave32):
//   A 16x32 f16: lanes 0-15 row M=lane, K in {0..7}u{16..23}; lanes 16-31
//                row M=lane-16, K in {8..15}u{24..31}.
//   B 32x16 f16: lanes 0-15 N=lane, K in {0..7}u{16..23} (2 per VGPR); lanes 16-31 K in {8..15}u{24..31}.
//   C 16x16 f32: VGPR r -> M=r (lanes 0-15) / M=8+r (lanes 16-31), N=lane%16.
// Bias is column-only => folded into the accumulator init (saves 32 VALU/tile).
// ---------------------------------------------------------------------------
__global__ __launch_bounds__(256) void k_edge_wmma(
    const _Float16* __restrict__ emb_h,
    const int* __restrict__ e_src, const int* __restrict__ e_dst,
    const float* __restrict__ W_msg, const float* __restrict__ b_msg,
    float* __restrict__ agg, int nTiles)
{
    __shared__ __align__(16) _Float16 sB[4 * 4 * 32 * SBP];  // 24.0 KB
    __shared__ __align__(16) _Float16 sA[8][16 * SAP];       // 34.0 KB
    __shared__ int sDst[8][16];
    __shared__ int sSrc[8][16];

    const int tid  = threadIdx.x;
    const int w    = tid >> 5;
    const int lane = tid & 31;

    // Stage W_msg into LDS in B-fragment-native order (f32 -> f16), once.
    for (int idx = tid; idx < 4 * 4 * 32 * 16; idx += 256) {
        const int h  = idx & 15;
        const int l  = (idx >> 4) & 31;
        const int nt = (idx >> 9) & 3;
        const int kc = idx >> 11;
        const int n  = nt * 16 + (l & 15);
        const int k  = kc * 32 + ((h < 8) ? h : h + 8) + ((l < 16) ? 0 : 8);
        sB[((kc * 4 + nt) * 32 + l) * SBP + h] = (_Float16)W_msg[k * 64 + n];
    }

    const int waves = (int)gridDim.x * 8;
    const int iters = (nTiles + waves - 1) / waves;     // uniform across grid
    const int ncol  = lane & 15;
    const int rbase = (lane < 16) ? 0 : 8;
    const int row   = lane & 15;

    // Per-lane output column is tile-invariant: preload the 4 biases once.
    float biasr[4];
    #pragma unroll
    for (int nt = 0; nt < 4; nt++) biasr[nt] = b_msg[nt * 16 + ncol];

    for (int it = 0; it < iters; ++it) {
        const int  tile   = it * waves + (int)blockIdx.x * 8 + w;
        const bool active = tile < nTiles;               // wave-uniform
        const int  base   = tile * 16;

        if (active) {
            if (lane < 16) sDst[w][lane]      = e_dst[base + lane];
            else           sSrc[w][lane - 16] = e_src[base + lane - 16];
            // Gather 16 edges x 64 feats of xi and (xj-xi) into the A tile.
            // (sDst/sSrc/sA are wave-private; per-wave DS ordering suffices.)
            for (int j = lane; j < 256; j += 32) {
                const int e = j >> 4;
                const int f = (j & 15) * 4;
                const h2* pi = (const h2*)(emb_h + (size_t)sDst[w][e] * HDIM + f);
                const h2* pj = (const h2*)(emb_h + (size_t)sSrc[w][e] * HDIM + f);
                h2 i0 = pi[0], i1 = pi[1];
                h2 j0 = pj[0], j1 = pj[1];
                *(h2*)&sA[w][e * SAP + f]          = i0;
                *(h2*)&sA[w][e * SAP + f + 2]      = i1;
                *(h2*)&sA[w][e * SAP + 64 + f]     = j0 - i0;
                *(h2*)&sA[w][e * SAP + 64 + f + 2] = j1 - i1;
            }
        }
        __syncthreads();   // covers sB on it==0; keeps block in lockstep after

        if (active) {
            // Load the four K-chunk A fragments.
            v16h afr[4];
            #pragma unroll
            for (int c = 0; c < 4; c++) {
                const int b1 = c * 32 + ((lane < 16) ? 0 : 8);
                const v8h lo = *(const v8h*)&sA[w][row * SAP + b1];
                const v8h hi = *(const v8h*)&sA[w][row * SAP + b1 + 16];
                #pragma unroll
                for (int i = 0; i < 8; i++) { afr[c][i] = lo[i]; afr[c][8 + i] = hi[i]; }
            }
            // Hoist atomic destinations into registers BEFORE any asm with a
            // "memory" clobber, so they are read from LDS exactly once.
            float* aptr[8];
            #pragma unroll
            for (int r = 0; r < 8; r++) {
                const int node = sDst[w][rbase + r];
                aptr[r] = agg + (size_t)node * HDIM + ncol;
            }
            #pragma unroll
            for (int nt = 0; nt < 4; nt++) {
                // Seed the accumulator with the bias (same column for all 8
                // C VGPRs) instead of zero: D = A*B + bias.
                v8f acc;
                #pragma unroll
                for (int i = 0; i < 8; i++) acc[i] = biasr[nt];
                #pragma unroll
                for (int kc = 0; kc < 4; kc++) {
                    const v8h* q = (const v8h*)&sB[((kc * 4 + nt) * 32 + lane) * SBP];
                    const v8h lo = q[0], hi = q[1];
                    v16h bfr;
                    #pragma unroll
                    for (int i = 0; i < 8; i++) { bfr[i] = lo[i]; bfr[8 + i] = hi[i]; }
                    acc = __builtin_amdgcn_wmma_f32_16x16x32_f16(
                        false, afr[kc], false, bfr, (short)0, acc, false, false);
                }
                #pragma unroll
                for (int r = 0; r < 8; r++) {
                    atomicMaxFOff(aptr[r], acc[r], nt * 16 * 4);
                }
            }
        }
    }
}

// Scalar fallback for E % 16 leftover edges (f32 path).
__global__ void k_edge_rem(
    const float* __restrict__ emb,
    const int* __restrict__ e_src, const int* __restrict__ e_dst,
    const float* __restrict__ W_msg, const float* __restrict__ b_msg,
    float* __restrict__ agg, int m)
{
    const int e = threadIdx.x;
    if (e >= m) return;
    const int si = e_src[e], di = e_dst[e];
    const float* xi = emb + (size_t)di * HDIM;
    const float* xj = emb + (size_t)si * HDIM;
    for (int n = 0; n < 64; n++) {
        float s = b_msg[n];
        for (int k = 0; k < 64; k++) {
            const float a = xi[k];
            s += a * W_msg[k * 64 + n] + (xj[k] - a) * W_msg[(64 + k) * 64 + n];
        }
        atomicMaxFOff(&agg[(size_t)di * HDIM + n], s, 0);
    }
}

// ---------------------------------------------------------------------------
// Kernel 4: emb += bn_d(where(isfinite(agg), agg, 0)); refresh f16 mirror.
// Vectorized x4 (b128 loads/stores).
// ---------------------------------------------------------------------------
__global__ __launch_bounds__(256) void k_apply(
    const float4* __restrict__ agg4,
    const float4* __restrict__ g4, const float4* __restrict__ b4,
    const float4* __restrict__ m4, const float4* __restrict__ v4,
    float4* __restrict__ emb4, h4* __restrict__ embh4, size_t total4)
{
    const size_t idx = (size_t)blockIdx.x * 256 + threadIdx.x;
    if (idx >= total4) return;
    const int q = (int)(idx & 15);          // float4 index within 64-feat row
    float4 a = agg4[idx];
    const float4 g = g4[q], b = b4[q], m = m4[q], v = v4[q];
    float4 e = emb4[idx];
    float r[4];
    const float av[4] = {a.x, a.y, a.z, a.w};
    const float gv[4] = {g.x, g.y, g.z, g.w};
    const float bv[4] = {b.x, b.y, b.z, b.w};
    const float mv[4] = {m.x, m.y, m.z, m.w};
    const float vv[4] = {v.x, v.y, v.z, v.w};
    const float ev[4] = {e.x, e.y, e.z, e.w};
    h4 hh;
    #pragma unroll
    for (int i = 0; i < 4; i++) {
        float aa = av[i];
        if (!(aa > -3.0e38f && aa < 3.0e38f)) aa = 0.0f;  // empty segs: -inf
        r[i] = ev[i] + (aa - mv[i]) * rsqrtf(vv[i] + EPSV) * gv[i] + bv[i];
        hh[i] = (_Float16)r[i];
    }
    float4 o; o.x = r[0]; o.y = r[1]; o.z = r[2]; o.w = r[3];
    emb4[idx]  = o;
    embh4[idx] = hh;
}

// ---------------------------------------------------------------------------
// Kernel 5: readout  out = elu(emb @ W1 + b1) @ W2 + b2
// ---------------------------------------------------------------------------
__global__ __launch_bounds__(256) void k_out(
    const float* __restrict__ emb,
    const float* __restrict__ W1, const float* __restrict__ b1,
    const float* __restrict__ W2, const float* __restrict__ b2,
    float* __restrict__ out, int N)
{
    __shared__ float sW1[64 * 32], sb1[32], sW2[32], sb2;
    const int tid = threadIdx.x;
    for (int i = tid; i < 64 * 32; i += 256) sW1[i] = W1[i];
    if (tid < 32) { sb1[tid] = b1[tid]; sW2[tid] = W2[tid]; }
    if (tid == 0) sb2 = b2[0];
    __syncthreads();

    const int n = blockIdx.x * 256 + tid;
    if (n >= N) return;
    float e[64];
    #pragma unroll
    for (int k = 0; k < 64; k++) e[k] = emb[(size_t)n * HDIM + k];
    float o = sb2;
    #pragma unroll
    for (int j = 0; j < 32; j++) {
        float s = sb1[j];
        #pragma unroll
        for (int k = 0; k < 64; k++) s += e[k] * sW1[k * 32 + j];
        o += eluf(s) * sW2[j];
    }
    out[n] = o;
}

// ---------------------------------------------------------------------------
extern "C" void kernel_launch(void* const* d_in, const int* in_sizes, int n_in,
                              void* d_out, int out_size, void* d_ws, size_t ws_size,
                              hipStream_t stream)
{
    const float* x_cont     = (const float*)d_in[0];
    const int*   x_cat      = (const int*)  d_in[1];
    const int*   edge_index = (const int*)  d_in[2];
    // d_in[3] = batch (unused)
    const float* datanorm   = (const float*)d_in[4];
    const float* emb_charge = (const float*)d_in[5];
    const float* emb_pdg    = (const float*)d_in[6];
    const float* W_cont     = (const float*)d_in[7];
    const float* b_cont     = (const float*)d_in[8];
    const float* W_cat      = (const float*)d_in[9];
    const float* b_cat      = (const float*)d_in[10];
    const float* W_enc      = (const float*)d_in[11];
    const float* b_enc      = (const float*)d_in[12];
    const float* bn0_g      = (const float*)d_in[13];
    const float* bn0_b      = (const float*)d_in[14];
    const float* bn0_m      = (const float*)d_in[15];
    const float* bn0_v      = (const float*)d_in[16];
    const float* W_msg      = (const float*)d_in[17];
    const float* b_msg      = (const float*)d_in[18];
    const float* bn_g       = (const float*)d_in[19];
    const float* bn_b       = (const float*)d_in[20];
    const float* bn_m       = (const float*)d_in[21];
    const float* bn_v       = (const float*)d_in[22];
    const float* W_out1     = (const float*)d_in[23];
    const float* b_out1     = (const float*)d_in[24];
    const float* W_out2     = (const float*)d_in[25];
    const float* b_out2     = (const float*)d_in[26];

    const int N = in_sizes[0] / 8;   // CD = 8
    const int E = in_sizes[2] / 2;

    // Workspace partition: emb f32 | emb f16 | agg f32
    const size_t nh = (size_t)N * HDIM;
    char* ws = (char*)d_ws;
    float*    emb   = (float*)ws;
    size_t    off   = (nh * sizeof(float) + 255) & ~(size_t)255;
    _Float16* emb_h = (_Float16*)(ws + off);
    off += (nh * sizeof(_Float16) + 255) & ~(size_t)255;
    float*    agg   = (float*)(ws + off);

    const int* e_src = edge_index;        // edge_index[0, :]
    const int* e_dst = edge_index + E;    // edge_index[1, :]

    // 1) encode
    k_encode<<<(N + 255) / 256, 256, 0, stream>>>(
        x_cont, x_cat, datanorm, emb_charge, emb_pdg,
        W_cont, b_cont, W_cat, b_cat, W_enc, b_enc,
        bn0_g, bn0_b, bn0_m, bn0_v, emb, emb_h, N);

    // 2) message passing, DEPTH = 2
    const int nTiles  = E / 16;
    const int rem     = E - nTiles * 16;
    int eBlocks = (nTiles + 7) / 8;
    if (eBlocks > 1536) eBlocks = 1536;    // persistent grid, ~9 tiles/wave
    const size_t n4   = nh / 4;
    const int    blk4 = (int)((n4 + 255) / 256);

    for (int d = 0; d < 2; d++) {
        k_fill_neginf<<<blk4, 256, 0, stream>>>((float4*)agg, n4);
        if (nTiles > 0)
            k_edge_wmma<<<eBlocks, 256, 0, stream>>>(
                emb_h, e_src, e_dst,
                W_msg + (size_t)d * 128 * 64, b_msg + d * 64, agg, nTiles);
        if (rem > 0)
            k_edge_rem<<<1, rem, 0, stream>>>(
                emb, e_src + nTiles * 16, e_dst + nTiles * 16,
                W_msg + (size_t)d * 128 * 64, b_msg + d * 64, agg, rem);
        k_apply<<<blk4, 256, 0, stream>>>(
            (const float4*)agg,
            (const float4*)(bn_g + d * 64), (const float4*)(bn_b + d * 64),
            (const float4*)(bn_m + d * 64), (const float4*)(bn_v + d * 64),
            (float4*)emb, (h4*)emb_h, n4);
    }

    // 3) readout
    k_out<<<(N + 255) / 256, 256, 0, stream>>>(
        emb, W_out1, b_out1, W_out2, b_out2, (float*)d_out, N);
}
